// TLPGN_56032143343772
// MI455X (gfx1250) — matrix operand order repaired and checked by reading, not compiled, not run on hardware
//
#include <hip/hip_runtime.h>
#include <hip/hip_bf16.h>
#include <math.h>

// ---------------------------------------------------------------------------
// TLPGN fused kernel for MI455X (gfx1250, wave32, WMMA bf16 -> f32, TDM)
//   B=128, N=96, D=128, M=8192, top-k=48 over the N axis per (b,m) column.
// Flash-attention style: Q resident in LDS, V streamed in 128-row chunks via
// the Tensor Data Mover (double-buffered tensor_load_to_lds), S recomputed in
// pass 2 (cheaper than 402MB of HBM attn traffic at 23.3 TB/s).
// ---------------------------------------------------------------------------

typedef __bf16 bf16_t;
typedef __bf16 v16bf __attribute__((ext_vector_type(16)));
typedef float  v8f   __attribute__((ext_vector_type(8)));
typedef unsigned int u32x4 __attribute__((ext_vector_type(4)));
typedef unsigned int u32x8 __attribute__((ext_vector_type(8)));

#define B_       128
#define N_       96
#define D_       128
#define M_       8192
#define KTOP     48
#define MT       128      // m-chunk width
#define STP      132      // padded LDS stride for S tile (conflict-free column scans)
#define NTHREADS 256      // 8 wave32 per workgroup
#define SPLIT    4        // M split for kernel 2 (512 blocks total)
#define SCALE    0.08838834764831845f  // 1/sqrt(128)

// LDS partition offsets (bytes)
#define K1_QS 0
#define K1_VS 24576
#define K1_ST (24576 + 32768)
#define K1_VF (K1_ST + 50688)                 // 2 x 65536 f32 staging
#define K1_SMEM (K1_VF + 131072)              // 239104

#define K2_QS 0
#define K2_VS 24576
#define K2_VT (24576 + 32768)
#define K2_ST (K2_VT + 32768)
#define K2_PS (K2_ST + 50688)
#define K2_MX (K2_PS + 24576)
#define K2_RI (K2_MX + 384)
#define K2_VF (K2_RI + 384)                   // 2 x 65536 f32 staging
#define K2_SMEM (K2_VF + 131072)              // 297216

// ---------------------------------------------------------------------------
// Tensor Data Mover: issue a 2D 128x128 f32 tile load Global -> LDS.
// D# groups 0/1 built in SGPRs (wave-uniform); VADDR2/3 omitted (<=2D tensor).
// Tracked with TENSORcnt; completion via s_wait_tensorcnt.
// ---------------------------------------------------------------------------
__device__ __forceinline__ void tdm_load_tile_f32_128x128(const float* gsrc,
                                                          unsigned int lds_off) {
  unsigned long long ga = (unsigned long long)(size_t)gsrc;
  u32x4 g0;
  g0[0] = 1u;                                        // count=1, user descriptor
  g0[1] = lds_off;                                   // LDS byte address
  g0[2] = (unsigned int)ga;                          // global_addr[31:0]
  g0[3] = (unsigned int)((ga >> 32) & 0x01FFFFFFu)   // global_addr[56:32]
        | (2u << 30);                                // type = 2 ("image")
  u32x8 g1;
  g1[0] = (2u << 16);        // workgroup_mask=0, data_size=2 (4B), no iter/pad
  g1[1] = (128u << 16);      // tensor_dim0[15:0]=128 (bits 63:48)
  g1[2] = (128u << 16);      // tensor_dim0 hi=0; tensor_dim1[15:0]=128
  g1[3] = (128u << 16);      // tensor_dim1 hi=0; tile_dim0=128
  g1[4] = 128u;              // tile_dim1=128; tile_dim2=0
  g1[5] = 128u;              // tensor_dim0_stride[31:0]=128 elements
  g1[6] = 0u;                // stride hi, tensor_dim1_stride lo
  g1[7] = 0u;
  asm volatile("tensor_load_to_lds %0, %1" : : "s"(g0), "s"(g1) : "memory");
}

// ---- WMMA fragment helpers (gfx1250 wave32 layouts, 16-bit A/B, f32 C/D) ----

// A (16x32, row-major source): lane%16 = row, lanes>=16 shift K by +16.
__device__ __forceinline__ v16bf load_A(const bf16_t* base, int ld, int r0, int k0, int lane) {
  v16bf a;
  const bf16_t* p = base + (r0 + (lane & 15)) * ld + k0 + ((lane & 16) ? 16 : 0);
#pragma unroll
  for (int v = 0; v < 8; ++v) {
    int off = 2 * v;
    a[2 * v]     = p[off];
    a[2 * v + 1] = p[off + 1];
  }
  return a;
}

// B (32x16) with B[k][col] = SRC[c0+col][k0+k]  (SRC row-major, contiguous in k)
__device__ __forceinline__ v16bf load_Bt(const bf16_t* base, int ld, int c0, int k0, int lane) {
  v16bf b;
  const bf16_t* p = base + (c0 + (lane & 15)) * ld + k0 + ((lane & 16) ? 16 : 0);
#pragma unroll
  for (int v = 0; v < 8; ++v) { b[2 * v] = p[2 * v]; b[2 * v + 1] = p[2 * v + 1]; }
  return b;
}

// C/D (16x16 f32): lane%16 = col, VGPR r = row r (+8 for lanes>=16)
__device__ __forceinline__ void store_C(float* base, int ld, int r0, int c0, v8f c, int lane) {
  float* p = base + (r0 + ((lane & 16) ? 8 : 0)) * ld + c0 + (lane & 15);
#pragma unroll
  for (int r = 0; r < 8; ++r) p[r * ld] = c[r];
}

// ---------------------------------------------------------------------------
// Kernel 1: per-batch online-softmax row stats (max, sumexp) + out = query
// ---------------------------------------------------------------------------
__global__ __launch_bounds__(NTHREADS)
void tlpgn_stats_kernel(const float* __restrict__ logits,
                        const float* __restrict__ tlv,
                        float* __restrict__ out,
                        float* __restrict__ stats) {
  extern __shared__ char smem[];
  bf16_t* qs = (bf16_t*)(smem + K1_QS);   // 96*128 bf16
  bf16_t* vs = (bf16_t*)(smem + K1_VS);   // 128*128 bf16
  float*  st = (float*)(smem + K1_ST);    // 96*132 f32
  float*  vf = (float*)(smem + K1_VF);    // 2 x 128*128 f32 (TDM landing)

  const int b    = blockIdx.x;
  const int tid  = threadIdx.x;
  const int wave = tid >> 5;
  const int lane = tid & 31;
  const unsigned int vf_off[2] = { (unsigned int)(size_t)(smem + K1_VF),
                                   (unsigned int)(size_t)(smem + K1_VF + 65536) };

  // prologue: kick off TDM for chunk 0 (wave 0 only; TDM ignores EXEC)
  if (wave == 0) tdm_load_tile_f32_128x128(tlv, vf_off[0]);

  for (int i = tid; i < N_ * D_; i += NTHREADS) {
    float q = logits[(size_t)b * N_ * D_ + i];
    qs[i] = (bf16_t)q;
    out[(size_t)b * N_ * D_ + i] = q;   // initialize output with query
  }

  float run_max = -1e30f, run_sum = 0.f;

  for (int ci = 0, m0 = 0; m0 < M_; m0 += MT, ++ci) {
    const int buf = ci & 1;
    if (wave == 0) {
      if (m0 + MT < M_) {   // prefetch next chunk into the other buffer
        tdm_load_tile_f32_128x128(tlv + (size_t)(m0 + MT) * D_, vf_off[1 - buf]);
        __builtin_amdgcn_s_wait_tensorcnt(1);   // current chunk landed
      } else {
        __builtin_amdgcn_s_wait_tensorcnt(0);
      }
    }
    __syncthreads();   // S1: vf[buf] visible to all waves

    // convert f32 chunk -> bf16 (LDS -> LDS)
    {
      const float4* v4 = (const float4*)(vf + (size_t)buf * MT * D_);
      for (int i = tid; i < MT * D_ / 4; i += NTHREADS) {
        float4 v = v4[i]; int o = i * 4;
        vs[o] = (bf16_t)v.x; vs[o + 1] = (bf16_t)v.y;
        vs[o + 2] = (bf16_t)v.z; vs[o + 3] = (bf16_t)v.w;
      }
    }
    __syncthreads();   // S2: vs ready

    // S = (Q * V^T) * scale : 6x8 tiles of 16x16, 6 tiles per wave
#pragma unroll
    for (int i = 0; i < 6; ++i) {
      int t = wave * 6 + i, nt = t >> 3, mt = t & 7;
      v8f c = {};
#pragma unroll
      for (int k = 0; k < D_; k += 32) {
        v16bf a  = load_A (qs, D_, nt * 16, k, lane);
        v16bf bb = load_Bt(vs, D_, mt * 16, k, lane);
        c = __builtin_amdgcn_wmma_f32_16x16x32_bf16(false, a, false, bb, (short)0, c, false, false);
      }
#pragma unroll
      for (int r = 0; r < 8; ++r) c[r] *= SCALE;
      store_C(st, STP, nt * 16, mt * 16, c, lane);
    }
    __syncthreads();

    // online softmax row stats over this chunk
    if (tid < N_) {
      float cm = -1e30f;
      for (int c = 0; c < MT; ++c) cm = fmaxf(cm, st[tid * STP + c]);
      float nm = fmaxf(run_max, cm);
      float s = 0.f;
      for (int c = 0; c < MT; ++c) s += __expf(st[tid * STP + c] - nm);
      run_sum = run_sum * __expf(run_max - nm) + s;
      run_max = nm;
    }
    __syncthreads();
  }

  if (tid < N_) {
    stats[((size_t)b * N_ + tid) * 2 + 0] = run_max;
    stats[((size_t)b * N_ + tid) * 2 + 1] = run_sum;
  }
}

// ---------------------------------------------------------------------------
// Kernel 2: recompute S, normalize, per-column top-48 mask, O += P*V (atomic)
// ---------------------------------------------------------------------------
__global__ __launch_bounds__(NTHREADS)
void tlpgn_main_kernel(const float* __restrict__ logits,
                       const float* __restrict__ tlv,
                       float* __restrict__ out,
                       const float* __restrict__ stats) {
  extern __shared__ char smem[];
  bf16_t* qs = (bf16_t*)(smem + K2_QS);   // Q bf16 (96x128)
  bf16_t* vs = (bf16_t*)(smem + K2_VS);   // V bf16 (128x128), row-major [m][d]
  bf16_t* vt = (bf16_t*)(smem + K2_VT);   // V bf16 transposed [d][m] for GEMM2 B-frags
  float*  st = (float*)(smem + K2_ST);    // S tile f32 (96 x STP)
  bf16_t* ps = (bf16_t*)(smem + K2_PS);   // masked P bf16 (96x128)
  float*  mx = (float*)(smem + K2_MX);    // row max (96)
  float*  ri = (float*)(smem + K2_RI);    // 1/rowsum (96)
  float*  vf = (float*)(smem + K2_VF);    // 2 x 128*128 f32 (TDM landing)

  const int b     = blockIdx.y;
  const int split = blockIdx.x;
  const int tid   = threadIdx.x;
  const int wave  = tid >> 5;
  const int lane  = tid & 31;
  const unsigned int vf_off[2] = { (unsigned int)(size_t)(smem + K2_VF),
                                   (unsigned int)(size_t)(smem + K2_VF + 65536) };

  const int mbeg = split * (M_ / SPLIT);
  const int mend = mbeg + (M_ / SPLIT);

  if (wave == 0) tdm_load_tile_f32_128x128(tlv + (size_t)mbeg * D_, vf_off[0]);

  for (int i = tid; i < N_ * D_; i += NTHREADS)
    qs[i] = (bf16_t)logits[(size_t)b * N_ * D_ + i];
  if (tid < N_) {
    mx[tid] = stats[((size_t)b * N_ + tid) * 2 + 0];
    ri[tid] = 1.0f / stats[((size_t)b * N_ + tid) * 2 + 1];
  }

  v8f acc[6] = {};   // persistent O accumulators: 6 tiles of 16x16 per wave

  for (int ci = 0, m0 = mbeg; m0 < mend; m0 += MT, ++ci) {
    const int buf = ci & 1;
    if (wave == 0) {
      if (m0 + MT < mend) {
        tdm_load_tile_f32_128x128(tlv + (size_t)(m0 + MT) * D_, vf_off[1 - buf]);
        __builtin_amdgcn_s_wait_tensorcnt(1);
      } else {
        __builtin_amdgcn_s_wait_tensorcnt(0);
      }
    }
    __syncthreads();   // S1: vf[buf] landed; prev-iter consumers of vs/vt/ps done

    // convert f32 chunk -> bf16, writing both row-major (vs) and transposed (vt)
    {
      const float4* v4 = (const float4*)(vf + (size_t)buf * MT * D_);
      for (int i = tid; i < MT * D_ / 4; i += NTHREADS) {
        float4 v = v4[i]; int o = i * 4;
        int m = o >> 7, d = o & (D_ - 1);
        bf16_t x = (bf16_t)v.x, y = (bf16_t)v.y, z = (bf16_t)v.z, w = (bf16_t)v.w;
        vs[o] = x; vs[o + 1] = y; vs[o + 2] = z; vs[o + 3] = w;
        vt[(d + 0) * MT + m] = x; vt[(d + 1) * MT + m] = y;
        vt[(d + 2) * MT + m] = z; vt[(d + 3) * MT + m] = w;
      }
    }
    __syncthreads();   // S2

    // ---- GEMM1: S = (Q * V^T) * scale ----
#pragma unroll
    for (int i = 0; i < 6; ++i) {
      int t = wave * 6 + i, nt = t >> 3, mt = t & 7;
      v8f c = {};
#pragma unroll
      for (int k = 0; k < D_; k += 32) {
        v16bf a  = load_A (qs, D_, nt * 16, k, lane);
        v16bf bb = load_Bt(vs, D_, mt * 16, k, lane);
        c = __builtin_amdgcn_wmma_f32_16x16x32_bf16(false, a, false, bb, (short)0, c, false, false);
      }
#pragma unroll
      for (int r = 0; r < 8; ++r) c[r] *= SCALE;
      store_C(st, STP, nt * 16, mt * 16, c, lane);
    }
    __syncthreads();

    // ---- normalize: attn = exp(s - rowmax) / rowsum ----
    for (int i = tid; i < N_ * MT; i += NTHREADS) {
      int n = i >> 7, c = i & (MT - 1);
      int idx = n * STP + c;
      st[idx] = __expf(st[idx] - mx[n]) * ri[n];
    }
    __syncthreads();

    // ---- per-column top-48 over N=96: 31-bit radix select on float bits ----
    if (tid < MT) {
      unsigned thr = 0u;
#pragma unroll 1
      for (int bit = 30; bit >= 0; --bit) {
        unsigned cand = thr | (1u << bit);
        int cnt = 0;
        for (int n = 0; n < N_; ++n)
          cnt += (__float_as_uint(st[n * STP + tid]) >= cand) ? 1 : 0;
        if (cnt >= KTOP) thr = cand;
      }
      float tval = __uint_as_float(thr);
      for (int n = 0; n < N_; ++n) {
        float v = st[n * STP + tid];
        ps[n * MT + tid] = (bf16_t)((v >= tval) ? v : 0.0f);
      }
    }
    __syncthreads();

    // ---- GEMM2: O += P * V  (B-frags contiguous from transposed vt) ----
#pragma unroll
    for (int i = 0; i < 6; ++i) {
      int t = wave * 6 + i, nt = t >> 3, dt = t & 7;
      v8f c = acc[i];
#pragma unroll
      for (int k = 0; k < MT; k += 32) {
        v16bf a  = load_A (ps, MT, nt * 16, k, lane);
        v16bf bb = load_Bt(vt, MT, dt * 16, k, lane);   // B[k][col]=V[k][dt*16+col]
        c = __builtin_amdgcn_wmma_f32_16x16x32_bf16(false, a, false, bb, (short)0, c, false, false);
      }
      acc[i] = c;
    }
  }

  // ---- accumulate into d_out (already holds query) ----
#pragma unroll
  for (int i = 0; i < 6; ++i) {
    int t = wave * 6 + i, nt = t >> 3, dt = t & 7;
    int col = lane & 15, rb = (lane & 16) ? 8 : 0;
#pragma unroll
    for (int r = 0; r < 8; ++r)
      atomicAdd(&out[(size_t)b * N_ * D_ + (size_t)(nt * 16 + rb + r) * D_ + dt * 16 + col],
                acc[i][r]);
  }
}

// ---------------------------------------------------------------------------
extern "C" void kernel_launch(void* const* d_in, const int* in_sizes, int n_in,
                              void* d_out, int out_size, void* d_ws, size_t ws_size,
                              hipStream_t stream) {
  (void)in_sizes; (void)n_in; (void)out_size; (void)ws_size;
  const float* logits = (const float*)d_in[0];   // (B, N*D) f32
  const float* tlv    = (const float*)d_in[1];   // (M, D)   f32
  float*       out    = (float*)d_out;           // (B, N, D) f32
  float*       stats  = (float*)d_ws;            // (B, N, 2) f32 row stats

  tlpgn_stats_kernel<<<dim3(B_), dim3(NTHREADS), K1_SMEM, stream>>>(logits, tlv, out, stats);
  tlpgn_main_kernel<<<dim3(SPLIT, B_), dim3(NTHREADS), K2_SMEM, stream>>>(logits, tlv, out, stats);
}
